// NAOMIImputer_51118700757605
// MI455X (gfx1250) — matrix operand order repaired and checked by reading, not compiled, not run on hardware
//
#include <hip/hip_runtime.h>
#include <hip/hip_bf16.h>

// ---------------------------------------------------------------------------
// Types for CDNA5 WMMA (wave32, 16x16x32 bf16 -> f32)
// ---------------------------------------------------------------------------
typedef __attribute__((ext_vector_type(16))) __bf16          v16bf;
typedef __attribute__((ext_vector_type(8)))  float           v8f;
typedef __attribute__((ext_vector_type(8)))  unsigned short  us8;
typedef __attribute__((ext_vector_type(16))) unsigned short  us16;

__device__ __forceinline__ unsigned short f2bf(float f) {
    union { float f; unsigned int u; } c; c.f = f;
    unsigned int u = c.u;
    unsigned int r = u + 0x7FFFu + ((u >> 16) & 1u);   // round-to-nearest-even
    return (unsigned short)(r >> 16);
}
__device__ __forceinline__ float bf2f(unsigned short h) {
    union { unsigned int u; float f; } c; c.u = ((unsigned int)h) << 16;
    return c.f;
}

// A fragment: 16x32 bf16 tile, rows row0..row0+15 of row-major src (stride lda),
// K window [k0, k0+32). Contiguous k layout (same order as B):
//   lanes 0-15: M=lane,   elems 0..15 -> K=0..15
//   lanes16-31: M=lane-16,elems 0..15 -> K=16..31
__device__ __forceinline__ v16bf load_a_frag(const unsigned short* src, int lda,
                                             int row0, int k0) {
    int lane = threadIdx.x & 31;
    int m    = lane & 15;
    int hi   = lane >> 4;
    const unsigned short* p = src + (size_t)(row0 + m) * lda + k0 + hi * 16;
    union { us16 v; us8 h[2]; } u;
    u.h[0] = *(const us8*)(p);
    u.h[1] = *(const us8*)(p + 8);
    return __builtin_bit_cast(v16bf, u.v);
}

// B fragment: 32x16 bf16 tile where B(k,n) = W[n][k] (W row-major, stride ldb).
// Layout: lanes 0-15: N=lane, K=0..15 packed; lanes 16-31: N=lane-16, K=16..31.
__device__ __forceinline__ v16bf load_b_frag(const unsigned short* W, int ldb,
                                             int ncol0, int k0) {
    int lane = threadIdx.x & 31;
    int n    = lane & 15;
    int hi   = lane >> 4;
    const unsigned short* p = W + (size_t)(ncol0 + n) * ldb + k0 + hi * 16;
    us16 v = *(const us16*)p;
    return __builtin_bit_cast(v16bf, v);
}

__device__ __forceinline__ v8f wmma_bf16(v16bf a, v16bf b, v8f c) {
    return __builtin_amdgcn_wmma_f32_16x16x32_bf16(
        /*neg_a=*/false, a, /*neg_b=*/false, b,
        /*c_mod=*/(short)0, c, /*reuse_a=*/false, /*reuse_b=*/false);
}

// ---------------------------------------------------------------------------
// Problem constants
// ---------------------------------------------------------------------------
#define TT   200
#define BB   128
#define HH   1024
#define YDIM 44
#define DDIM 1024

// ---------------------------------------------------------------------------
// Conversion / init kernels
// ---------------------------------------------------------------------------
__global__ void k_zero_us(unsigned short* p, int n) {
    int i = blockIdx.x * blockDim.x + threadIdx.x;
    if (i < n) p[i] = 0;
}
__global__ void k_zero_f(float* p, int n) {
    int i = blockIdx.x * blockDim.x + threadIdx.x;
    if (i < n) p[i] = 0.f;
}
// dst[r][c<cols]=bf(src[r][c]); pad c in [cols,ldd) with 0.
__global__ void k_cvt_pad(const float* __restrict__ src, unsigned short* __restrict__ dst,
                          int rows, int cols, int ldd) {
    int i = blockIdx.x * blockDim.x + threadIdx.x;
    int total = rows * ldd;
    if (i >= total) return;
    int r = i / ldd, c = i - r * ldd;
    dst[i] = (c < cols) ? f2bf(src[(size_t)r * cols + c]) : (unsigned short)0;
}
// src R x C row-major  ->  dst Cpad x R  (dst[c][r] = src[r][c], rows c>=C zeroed)
__global__ void k_cvt_transpose(const float* __restrict__ src, unsigned short* __restrict__ dst,
                                int R, int C, int Cpad) {
    int i = blockIdx.x * blockDim.x + threadIdx.x;
    int total = Cpad * R;
    if (i >= total) return;
    int c = i / R, r = i - c * R;
    dst[i] = (c < C) ? f2bf(src[(size_t)r * C + c]) : (unsigned short)0;
}
// backward GRU input: step s reads data[199 - s]; pad 45 -> 64
__global__ void k_cvt_data_rev(const float* __restrict__ data, unsigned short* __restrict__ dst) {
    const int S = TT - 1, Din = YDIM + 1, Dp = 64;
    int i = blockIdx.x * blockDim.x + threadIdx.x;
    int total = S * BB * Dp;
    if (i >= total) return;
    int d = i % Dp;
    int b = (i / Dp) % BB;
    int s = i / (Dp * BB);
    dst[i] = (d < Din) ? f2bf(data[((size_t)(S - s) * BB + b) * Din + d])
                       : (unsigned short)0;
}

// ---------------------------------------------------------------------------
// Fused quad GRU step.  One launch advances four independent GRU chains by one
// time step each (fwd-L0, fwd-L1 one step behind, bwd-L0, bwd-L1 one step
// behind).  128 WGs: blockIdx.x>>5 selects the chain, low 5 bits select the
// 32-wide hidden-column block.  Inside a WG: 8 waves x 16-row batch strips.
// acc_r/acc_z accumulate gi+gh together; acc_in (x-path) and acc_hn (h-path)
// are kept separate for  n = tanh(gi_n + r * gh_n).
// ---------------------------------------------------------------------------
struct GruArgs {
    const unsigned short* x;        // B x Dpad bf16 (step slice)
    const unsigned short* h_in;     // B x H bf16
    unsigned short*       h_out;    // B x H bf16
    unsigned short*       out_seq;  // B x H bf16 (target seq slot)
    const unsigned short* Wih;      // 3H x Dpad bf16
    const unsigned short* Whh;      // 3H x H bf16
    const float*          bih;      // 3H f32
    const float*          bhh;      // 3H f32
    int Dpad;
    int valid;
};

__device__ __forceinline__ void gru_unit(const GruArgs& a, int j0) {
    int wave = threadIdx.x >> 5;
    int lane = threadIdx.x & 31;
    int row0 = wave * 16;

    v8f acc_r[2]  = {};
    v8f acc_z[2]  = {};
    v8f acc_in[2] = {};
    v8f acc_hn[2] = {};

    // phase 1: gi = x @ Wih^T
    for (int k0 = 0; k0 < a.Dpad; k0 += 32) {
        v16bf af = load_a_frag(a.x, a.Dpad, row0, k0);
        #pragma unroll
        for (int tc = 0; tc < 2; ++tc) {
            int jc = j0 + tc * 16;
            v16bf br = load_b_frag(a.Wih, a.Dpad, jc,          k0);
            v16bf bz = load_b_frag(a.Wih, a.Dpad, jc + HH,     k0);
            v16bf bn = load_b_frag(a.Wih, a.Dpad, jc + 2 * HH, k0);
            acc_r[tc]  = wmma_bf16(af, br, acc_r[tc]);
            acc_z[tc]  = wmma_bf16(af, bz, acc_z[tc]);
            acc_in[tc] = wmma_bf16(af, bn, acc_in[tc]);
        }
    }
    // phase 2: gh = h @ Whh^T
    for (int k0 = 0; k0 < HH; k0 += 32) {
        v16bf af = load_a_frag(a.h_in, HH, row0, k0);
        #pragma unroll
        for (int tc = 0; tc < 2; ++tc) {
            int jc = j0 + tc * 16;
            v16bf br = load_b_frag(a.Whh, HH, jc,          k0);
            v16bf bz = load_b_frag(a.Whh, HH, jc + HH,     k0);
            v16bf bn = load_b_frag(a.Whh, HH, jc + 2 * HH, k0);
            acc_r[tc]  = wmma_bf16(af, br, acc_r[tc]);
            acc_z[tc]  = wmma_bf16(af, bz, acc_z[tc]);
            acc_hn[tc] = wmma_bf16(af, bn, acc_hn[tc]);
        }
    }

    // epilogue: gates + state update.  C layout: n=lane&15, m=v+8*(lane>>4)
    int hi = lane >> 4, nn = lane & 15;
    #pragma unroll
    for (int tc = 0; tc < 2; ++tc) {
        int j = j0 + tc * 16 + nn;
        float br_  = a.bih[j]          + a.bhh[j];
        float bz_  = a.bih[j + HH]     + a.bhh[j + HH];
        float bin_ = a.bih[j + 2 * HH];
        float bhn_ = a.bhh[j + 2 * HH];
        #pragma unroll
        for (int v = 0; v < 8; ++v) {
            int b  = row0 + v + 8 * hi;
            float r  = 1.f / (1.f + __expf(-(acc_r[tc][v] + br_)));
            float z  = 1.f / (1.f + __expf(-(acc_z[tc][v] + bz_)));
            float ng = tanhf(acc_in[tc][v] + bin_ + r * (acc_hn[tc][v] + bhn_));
            float hp = bf2f(a.h_in[(size_t)b * HH + j]);
            unsigned short hb = f2bf((1.f - z) * ng + z * hp);
            a.h_out[(size_t)b * HH + j]   = hb;
            a.out_seq[(size_t)b * HH + j] = hb;
        }
    }
}

__global__ __launch_bounds__(256)
void k_gru_quad(GruArgs a0, GruArgs a1, GruArgs a2, GruArgs a3) {
    int unit = blockIdx.x >> 5;
    int j0   = (blockIdx.x & 31) * 32;
    const GruArgs& a = (unit == 0) ? a0 : (unit == 1) ? a1 : (unit == 2) ? a2 : a3;
    if (!a.valid) return;          // uniform per-WG branch (EXEC stays all-1s)
    gru_unit(a, j0);
}

// ---------------------------------------------------------------------------
// Decoder GEMM1: dec = relu([Hf[t]; HB[2k+t]] @ Wd + bd)  -> bf16
// Grid: (DDIM/64, Tk), 256 threads. K = 2048.
// ---------------------------------------------------------------------------
__global__ __launch_bounds__(256)
void k_dec_gemm1(const unsigned short* __restrict__ Hf,
                 const unsigned short* __restrict__ HBk,   // HB + 2k*B*H
                 const unsigned short* __restrict__ WdT,   // DDIM x 2048 bf16
                 const float* __restrict__ bd,
                 unsigned short* __restrict__ dec) {
    int t    = blockIdx.y;
    int c0   = blockIdx.x * 64;
    int wave = threadIdx.x >> 5, lane = threadIdx.x & 31;
    int row0 = wave * 16;
    const unsigned short* Af = Hf  + (size_t)t * BB * HH;
    const unsigned short* Ab = HBk + (size_t)t * BB * HH;

    v8f acc[4] = {};
    for (int k0 = 0; k0 < 2 * HH; k0 += 32) {
        v16bf af = (k0 < HH) ? load_a_frag(Af, HH, row0, k0)
                             : load_a_frag(Ab, HH, row0, k0 - HH);
        #pragma unroll
        for (int tc = 0; tc < 4; ++tc) {
            v16bf bf = load_b_frag(WdT, 2 * HH, c0 + tc * 16, k0);
            acc[tc] = wmma_bf16(af, bf, acc[tc]);
        }
    }
    int hi = lane >> 4, nn = lane & 15;
    #pragma unroll
    for (int tc = 0; tc < 4; ++tc) {
        int c = c0 + tc * 16 + nn;
        float bias = bd[c];
        #pragma unroll
        for (int v = 0; v < 8; ++v) {
            int b = row0 + v + 8 * hi;
            float d = acc[tc][v] + bias;
            dec[((size_t)t * BB + b) * DDIM + c] = f2bf(d > 0.f ? d : 0.f);
        }
    }
}

// ---------------------------------------------------------------------------
// Decoder GEMM2 + loss: mean = dec @ Wm + bm; accumulate L1 and pair distance.
// Grid: (Tk), 256 threads. N padded 44 -> 48.
// ---------------------------------------------------------------------------
__global__ __launch_bounds__(256)
void k_dec_gemm2_loss(const unsigned short* __restrict__ dec,
                      const unsigned short* __restrict__ WmT,   // 48 x 1024 bf16
                      const float* __restrict__ bm,
                      const float* __restrict__ gt,
                      int kk, float* __restrict__ acc_out) {
    int t    = blockIdx.x;
    int wave = threadIdx.x >> 5, lane = threadIdx.x & 31;
    int row0 = wave * 16;
    const unsigned short* A = dec + (size_t)t * BB * DDIM;

    v8f acc[3] = {};
    for (int k0 = 0; k0 < DDIM; k0 += 32) {
        v16bf af = load_a_frag(A, DDIM, row0, k0);
        #pragma unroll
        for (int tc = 0; tc < 3; ++tc) {
            v16bf bf = load_b_frag(WmT, DDIM, tc * 16, k0);
            acc[tc] = wmma_bf16(af, bf, acc[tc]);
        }
    }
    int hi = lane >> 4, nn = lane & 15;
    float l1 = 0.f, pos = 0.f;
    const float* gts = gt + (size_t)(t + kk) * BB * YDIM;
    #pragma unroll
    for (int tc = 0; tc < 3; ++tc) {
        int y = tc * 16 + nn;
        bool valid = (y < YDIM);
        float bias  = valid ? bm[y] : 0.f;
        float scale = (y & 1) ? 72.f : 108.f;
        #pragma unroll
        for (int v = 0; v < 8; ++v) {
            int b = row0 + v + 8 * hi;
            float mean = acc[tc][v] + bias;
            float nxt  = valid ? gts[(size_t)b * YDIM + y] : 0.f;
            float diff = mean - nxt;
            if (valid) l1 += fabsf(diff);
            float sd    = diff * scale;
            float other = __shfl_xor(sd, 1, 32);   // partner coordinate (y^1)
            if (valid && ((y & 1) == 0)) pos += sqrtf(sd * sd + other * other);
        }
    }
    __shared__ float red0[256], red1[256];
    red0[threadIdx.x] = l1; red1[threadIdx.x] = pos;
    __syncthreads();
    for (int s = 128; s > 0; s >>= 1) {
        if ((int)threadIdx.x < s) {
            red0[threadIdx.x] += red0[threadIdx.x + s];
            red1[threadIdx.x] += red1[threadIdx.x + s];
        }
        __syncthreads();
    }
    if (threadIdx.x == 0) {
        atomicAdd(&acc_out[0], red0[0]);
        atomicAdd(&acc_out[1], red1[0]);
    }
}

__global__ void k_finalize(const float* __restrict__ acc, float* __restrict__ out) {
    if (threadIdx.x == 0 && blockIdx.x == 0) {
        // loss: sum / (B*YD) per t, summed over t,k, then / T / B
        out[0] = acc[0] / ((float)BB * (float)YDIM) / (float)TT / (float)BB;
        // pos: sum / (B*22) per t, summed, / imput (= 199+197+193+185 = 774)
        out[1] = acc[1] / ((float)BB * 22.f) / 774.f;
    }
}

// ---------------------------------------------------------------------------
// Host launcher
// ---------------------------------------------------------------------------
extern "C" void kernel_launch(void* const* d_in, const int* in_sizes, int n_in,
                              void* d_out, int out_size, void* d_ws, size_t ws_size,
                              hipStream_t stream) {
    (void)in_sizes; (void)n_in; (void)out_size; (void)ws_size;
    const float* data = (const float*)d_in[0];
    const float* gt   = (const float*)d_in[1];
    // gru_w: [Wih,Whh,bih,bhh] x 2 layers -> idx 2..9 ; back_w -> 10..17
    // dec_w: [Wd,bd,Wm,bm] for k=1,2,4,8 -> idx 18..33
    const float* fW[2][2] = {{(const float*)d_in[2],  (const float*)d_in[3]},
                             {(const float*)d_in[6],  (const float*)d_in[7]}};
    const float* fB[2][2] = {{(const float*)d_in[4],  (const float*)d_in[5]},
                             {(const float*)d_in[8],  (const float*)d_in[9]}};
    const float* bW[2][2] = {{(const float*)d_in[10], (const float*)d_in[11]},
                             {(const float*)d_in[14], (const float*)d_in[15]}};
    const float* bB[2][2] = {{(const float*)d_in[12], (const float*)d_in[13]},
                             {(const float*)d_in[16], (const float*)d_in[17]}};

    char* ws = (char*)d_ws; size_t off = 0;
    auto alloc = [&](size_t bytes) -> char* {
        char* p = ws + off;
        off += (bytes + 255) & ~(size_t)255;
        return p;
    };
    // activations (bf16 = unsigned short storage)
    unsigned short* gt_bf  = (unsigned short*)alloc((size_t)TT * BB * 64 * 2);
    unsigned short* drv_bf = (unsigned short*)alloc((size_t)(TT - 1) * BB * 64 * 2);
    unsigned short* seq_f  = (unsigned short*)alloc((size_t)TT * BB * HH * 2); // fwd L0 out; later dec scratch
    unsigned short* seq_b  = (unsigned short*)alloc((size_t)(TT - 1) * BB * HH * 2); // bwd L0 out
    unsigned short* Hf     = (unsigned short*)alloc((size_t)TT * BB * HH * 2);
    unsigned short* HB     = (unsigned short*)alloc((size_t)(TT + 1) * BB * HH * 2);
    unsigned short* hbufs  = (unsigned short*)alloc((size_t)8 * BB * HH * 2);  // 4 chains x ping/pong
    // weights bf16
    unsigned short* fWih[2], *fWhh[2], *bWih[2], *bWhh[2];
    fWih[0] = (unsigned short*)alloc((size_t)3 * HH * 64 * 2);
    fWhh[0] = (unsigned short*)alloc((size_t)3 * HH * HH * 2);
    fWih[1] = (unsigned short*)alloc((size_t)3 * HH * HH * 2);
    fWhh[1] = (unsigned short*)alloc((size_t)3 * HH * HH * 2);
    bWih[0] = (unsigned short*)alloc((size_t)3 * HH * 64 * 2);
    bWhh[0] = (unsigned short*)alloc((size_t)3 * HH * HH * 2);
    bWih[1] = (unsigned short*)alloc((size_t)3 * HH * HH * 2);
    bWhh[1] = (unsigned short*)alloc((size_t)3 * HH * HH * 2);
    unsigned short* WdT[4], *WmT[4];
    for (int i = 0; i < 4; ++i) {
        WdT[i] = (unsigned short*)alloc((size_t)DDIM * 2 * HH * 2);
        WmT[i] = (unsigned short*)alloc((size_t)48 * DDIM * 2);
    }
    float* accs = (float*)alloc(2 * sizeof(float));

    auto cvt = [&](const float* src, unsigned short* dst, int rows, int cols, int ldd) {
        int tot = rows * ldd;
        k_cvt_pad<<<(tot + 255) / 256, 256, 0, stream>>>(src, dst, rows, cols, ldd);
    };

    // ---- conversions & init ----
    k_zero_f<<<1, 32, 0, stream>>>(accs, 2);
    cvt(gt, gt_bf, TT * BB, YDIM, 64);
    {   int tot = (TT - 1) * BB * 64;
        k_cvt_data_rev<<<(tot + 255) / 256, 256, 0, stream>>>(data, drv_bf); }
    cvt(fW[0][0], fWih[0], 3 * HH, YDIM,     64);
    cvt(fW[0][1], fWhh[0], 3 * HH, HH,       HH);
    cvt(fW[1][0], fWih[1], 3 * HH, HH,       HH);
    cvt(fW[1][1], fWhh[1], 3 * HH, HH,       HH);
    cvt(bW[0][0], bWih[0], 3 * HH, YDIM + 1, 64);
    cvt(bW[0][1], bWhh[0], 3 * HH, HH,       HH);
    cvt(bW[1][0], bWih[1], 3 * HH, HH,       HH);
    cvt(bW[1][1], bWhh[1], 3 * HH, HH,       HH);
    for (int i = 0; i < 4; ++i) {
        int tot1 = DDIM * 2 * HH;
        k_cvt_transpose<<<(tot1 + 255) / 256, 256, 0, stream>>>(
            (const float*)d_in[18 + 4 * i], WdT[i], 2 * HH, DDIM, DDIM);
        int tot2 = 48 * DDIM;
        k_cvt_transpose<<<(tot2 + 255) / 256, 256, 0, stream>>>(
            (const float*)d_in[20 + 4 * i], WmT[i], DDIM, YDIM, 48);
    }
    // zero HB caps (t=0, t=T) and all hidden-state ping-pong buffers
    k_zero_us<<<(BB * HH + 255) / 256, 256, 0, stream>>>(HB, BB * HH);
    k_zero_us<<<(BB * HH + 255) / 256, 256, 0, stream>>>(HB + (size_t)TT * BB * HH, BB * HH);
    {   int tot = 8 * BB * HH;
        k_zero_us<<<(tot + 255) / 256, 256, 0, stream>>>(hbufs, tot); }

    // ---- pipelined quad-GRU: 201 dependent launches total ----
    unsigned short* hp[4][2];
    for (int c = 0; c < 4; ++c) {
        hp[c][0] = hbufs + (size_t)(2 * c)     * BB * HH;
        hp[c][1] = hbufs + (size_t)(2 * c + 1) * BB * HH;
    }
    int cur[4] = {0, 0, 0, 0};
    for (int t = 0; t <= TT; ++t) {
        GruArgs A[4] = {};
        bool run[4];
        // chain 0: forward layer 0, step t
        run[0] = (t <= TT - 1);
        if (run[0]) {
            A[0] = { gt_bf + (size_t)t * BB * 64, hp[0][cur[0]], hp[0][cur[0] ^ 1],
                     seq_f + (size_t)t * BB * HH,
                     fWih[0], fWhh[0], fB[0][0], fB[0][1], 64, 1 };
        }
        // chain 1: forward layer 1, step t-1
        run[1] = (t >= 1);
        if (run[1]) {
            int s = t - 1;
            A[1] = { seq_f + (size_t)s * BB * HH, hp[1][cur[1]], hp[1][cur[1] ^ 1],
                     Hf + (size_t)s * BB * HH,
                     fWih[1], fWhh[1], fB[1][0], fB[1][1], HH, 1 };
        }
        // chain 2: backward layer 0, step t
        run[2] = (t <= TT - 2);
        if (run[2]) {
            A[2] = { drv_bf + (size_t)t * BB * 64, hp[2][cur[2]], hp[2][cur[2] ^ 1],
                     seq_b + (size_t)t * BB * HH,
                     bWih[0], bWhh[0], bB[0][0], bB[0][1], 64, 1 };
        }
        // chain 3: backward layer 1, step t-1 -> HB[199-(t-1)]
        run[3] = (t >= 1 && t - 1 <= TT - 2);
        if (run[3]) {
            int s = t - 1;
            A[3] = { seq_b + (size_t)s * BB * HH, hp[3][cur[3]], hp[3][cur[3] ^ 1],
                     HB + (size_t)(TT - 1 - s) * BB * HH,
                     bWih[1], bWhh[1], bB[1][0], bB[1][1], HH, 1 };
        }
        k_gru_quad<<<128, 256, 0, stream>>>(A[0], A[1], A[2], A[3]);
        for (int c = 0; c < 4; ++c)
            if (run[c]) cur[c] ^= 1;
    }

    // ---- decoders (seq_f reused as `dec` scratch) ----
    const int ks[4] = {1, 2, 4, 8};
    for (int i = 0; i < 4; ++i) {
        int kk = ks[i];
        int Tk = TT - 2 * kk + 1;
        dim3 g1(DDIM / 64, Tk);
        k_dec_gemm1<<<g1, 256, 0, stream>>>(Hf, HB + (size_t)2 * kk * BB * HH,
                                            WdT[i], (const float*)d_in[19 + 4 * i], seq_f);
        k_dec_gemm2_loss<<<Tk, 256, 0, stream>>>(seq_f, WmT[i],
                                                 (const float*)d_in[21 + 4 * i],
                                                 gt, kk, accs);
    }
    k_finalize<<<1, 32, 0, stream>>>(accs, (float*)d_out);
}
